// EdgeConvEncoder_3341484556341
// MI455X (gfx1250) — matrix operand order repaired and checked by reading, not compiled, not run on hardware
//
#include <hip/hip_runtime.h>
#include <hip/hip_bf16.h>

// -------- CDNA5 WMMA types ----------------------------------------------
typedef __attribute__((ext_vector_type(16))) __bf16 v16bf;
typedef __attribute__((ext_vector_type(8)))  float  v8f;

__device__ __forceinline__ unsigned short f2bf(float f) {
  // round-to-nearest-even fp32 -> bf16 truncation
  unsigned int u = __float_as_uint(f);
  unsigned int r = u + 0x7FFFu + ((u >> 16) & 1u);
  return (unsigned short)(r >> 16);
}

// -------- pack weights (K x O row-major f32) into WMMA B-fragments -------
// B layout (16-bit, 32x16 KxN): VGPR j holds packed pair; lanes 0-15: N=lane,
// K = 2j,2j+1 (j<4), 2j+8,2j+9 (j>=4); lanes 16-31: N=lane-16, same K +8.
// Storage: frag[((kb*NT + nt)*32 + lane)*8 + j] as uint (bf16 pair).
__global__ void prep_wfrag(const float* __restrict__ w,
                           unsigned int* __restrict__ dst, int K, int O) {
  int i = blockIdx.x * blockDim.x + threadIdx.x;
  int total = (K * O) >> 1;
  if (i >= total) return;
  int NT   = O >> 4;
  int j    = i & 7;
  int lane = (i >> 3) & 31;
  int nt   = (i >> 8) % NT;
  int kb   = (i >> 8) / NT;
  int n    = nt * 16 + (lane & 15);
  int k0   = kb * 32 + (lane >> 4) * 8 + 2 * j + ((j >> 2) << 3);
  unsigned int lo = f2bf(w[(size_t)k0 * O + n]);
  unsigned int hi = f2bf(w[(size_t)(k0 + 1) * O + n]);
  dst[i] = lo | (hi << 16);
}

__global__ void zero_kernel(float* __restrict__ p, long n) {
  long i = (long)blockIdx.x * blockDim.x + threadIdx.x;
  if (i < n) p[i] = 0.0f;
}

__global__ void mean_relu_kernel(const float* __restrict__ s,
                                 const float* __restrict__ cnt,
                                 float* __restrict__ out, int O, long total) {
  long i = (long)blockIdx.x * blockDim.x + threadIdx.x;
  if (i >= total) return;
  float c = cnt[i / O];
  out[i] = fmaxf(s[i] / fmaxf(c, 1.0f), 0.0f);
}

__global__ void mean_add_relu_kernel(const float* __restrict__ s,
                                     const float* __restrict__ cnt,
                                     const float* __restrict__ x0,
                                     float* __restrict__ out, int O, long total) {
  long i = (long)blockIdx.x * blockDim.x + threadIdx.x;
  if (i >= total) return;
  float c = cnt[i / O];
  out[i] = fmaxf(s[i] / fmaxf(c, 1.0f) + x0[i], 0.0f);
}

// -------- fused EdgeConv layer kernel ------------------------------------
// Block: 128 threads = 4 wave32, handles 16 edges (one WMMA M-tile).
// K = feat dim (2*C), O = out dim. N_HEADS = 8 -> head size HS = O/8.
template <int K, int O>
__global__ __launch_bounds__(128, 2)
void edgeconv_kernel(const float* __restrict__ x,          // [N, K/2]
                     const int* __restrict__ esrc,
                     const int* __restrict__ edst,
                     const unsigned int* __restrict__ wfq, // packed B frags
                     const unsigned int* __restrict__ wfk,
                     const unsigned int* __restrict__ wfv,
                     const float* __restrict__ bq,
                     const float* __restrict__ bk,
                     const float* __restrict__ bv,
                     float* __restrict__ sacc,             // [N, O]
                     float* __restrict__ cnt,              // [N]
                     int nEdges) {
  constexpr int C   = K / 2;    // input channels
  constexpr int NT  = O / 16;   // N tiles
  constexpr int NPW = NT / 4;   // N tiles per wave (4 waves)
  constexpr int HS  = O / 8;    // head size == channels per phase-3 thread
  constexpr int KB  = K / 32;   // K blocks
  constexpr float SCALE = (HS == 16) ? 0.25f : 0.35355339059327373f;

  __shared__ unsigned short feat[16 * K];    // bf16 edge features
  __shared__ float qkv[3][16][O];            // f32 q/k/v tiles
  __shared__ int lsrc[16];
  __shared__ int ldst[16];

  const int t = threadIdx.x;
  const int ebase = blockIdx.x * 16;

  if (t < 16) {
    int e = ebase + t;
    lsrc[t] = (e < nEdges) ? esrc[e] : 0;
  } else if (t < 32) {
    int e = ebase + (t - 16);
    ldst[t - 16] = (e < nEdges) ? edst[e] : 0;
  }
  __syncthreads();

  // ---- phase 1: gather features into LDS as bf16: [xi | xj - xi] -------
  {
    const int em  = t >> 3;            // edge 0..15
    const int p   = t & 7;
    constexpr int CPP = C / 8;         // channels per thread (multiple of 2)
    const float* xi = x + (size_t)ldst[em] * C;
    const float* xj = x + (size_t)lsrc[em] * C;
    unsigned int* rowlo = (unsigned int*)&feat[em * K];
    unsigned int* rowhi = (unsigned int*)&feat[em * K + C];
#pragma unroll
    for (int c = p * CPP; c < p * CPP + CPP; c += 2) {
      float a0 = xi[c], a1 = xi[c + 1];
      float b0 = xj[c], b1 = xj[c + 1];
      rowlo[c >> 1] = (unsigned int)f2bf(a0) | ((unsigned int)f2bf(a1) << 16);
      rowhi[c >> 1] =
          (unsigned int)f2bf(b0 - a0) | ((unsigned int)f2bf(b1 - a1) << 16);
    }
  }
  __syncthreads();

  // ---- phase 2: WMMA GEMMs  q|k|v = feat @ W ----------------------------
  {
    const int wv_  = t >> 5;           // wave 0..3
    const int lane = t & 31;
    const int ntb  = wv_ * NPW;

    v8f acc[3][NPW];
#pragma unroll
    for (int m = 0; m < 3; ++m)
#pragma unroll
      for (int u = 0; u < NPW; ++u)
#pragma unroll
        for (int i = 0; i < 8; ++i) acc[m][u][i] = 0.0f;

    // A fragment (16x32 bf16): lanes 0-15 row=lane, K {0..7,16..23};
    // lanes 16-31 row=lane-16, K {8..15,24..31}.
    const int arow  = lane & 15;
    const int akoff = (lane >> 4) * 8;
    const unsigned int* const wfm[3] = {wfq, wfk, wfv};

#pragma unroll
    for (int kb = 0; kb < KB; ++kb) {
      union { uint4 q[2]; v16bf v; } afrag;
      const uint4* ap = (const uint4*)&feat[arow * K + kb * 32 + akoff];
      afrag.q[0] = ap[0];   // 8 bf16 (K+0..7 rel.)
      afrag.q[1] = ap[2];   // 8 bf16 (K+16..23 rel.) -> +32 bytes
#pragma unroll
      for (int m = 0; m < 3; ++m) {
#pragma unroll
        for (int u = 0; u < NPW; ++u) {
          const int nt = ntb + u;
          const uint4* bp =
              (const uint4*)(wfm[m] + ((size_t)(kb * NT + nt) * 32 + lane) * 8);
          union { uint4 q[2]; v16bf v; } bfrag;
          bfrag.q[0] = bp[0];
          bfrag.q[1] = bp[1];
          acc[m][u] = __builtin_amdgcn_wmma_f32_16x16x32_bf16(
              false, afrag.v, false, bfrag.v, (short)0, acc[m][u], false,
              false);
        }
      }
    }

    // spill C tiles to LDS. C layout: VGPR i -> M=i (lanes 0-15) / M=i+8.
    const int col0  = lane & 15;
    const int rbase = (lane >> 4) * 8;
#pragma unroll
    for (int m = 0; m < 3; ++m)
#pragma unroll
      for (int u = 0; u < NPW; ++u) {
        const int nt = ntb + u;
#pragma unroll
        for (int i = 0; i < 8; ++i)
          qkv[m][rbase + i][nt * 16 + col0] = acc[m][u][i];
      }
  }
  __syncthreads();

  // ---- phase 3: per-edge elementwise softmax over heads + scatter -------
  {
    const int em = t >> 3;
    const int p  = t & 7;        // one head per thread (HS channels)
    const int e  = ebase + em;
    if (e < nEdges) {
      const int dn = ldst[em];
      const int sn = lsrc[em];
      float dd = fabsf((float)(dn - sn));
      // sign(log d - log 8): d>8 -> +1, d<8 (incl. d=0 -> -inf) -> -1, d=8 -> 0
      float ew = (dd > 8.0f) ? 1.0f : ((dd < 8.0f) ? -1.0f : 0.0f);
      const int c0 = p * HS;
      float qv[HS], vv[HS];
#pragma unroll
      for (int j = 0; j < HS; ++j) {
        int c = c0 + j;
        float qq = qkv[0][em][c] + bq[c];
        float kk = qkv[1][em][c] + bk[c];
        vv[j] = (qkv[2][em][c] + bv[c]) * ew;
        qv[j] = qq * kk * SCALE;
      }
      float mx = -1e30f;
#pragma unroll
      for (int j = 0; j < HS; ++j) mx = fmaxf(mx, qv[j]);
      float sm = 0.0f;
#pragma unroll
      for (int j = 0; j < HS; ++j) { qv[j] = __expf(qv[j] - mx); sm += qv[j]; }
      float inv = 1.0f / sm;
      float* srow = sacc + (size_t)dn * O + c0;
#pragma unroll
      for (int j = 0; j < HS; ++j) atomicAdd(srow + j, qv[j] * inv * vv[j]);
      if (p == 0) atomicAdd(cnt + dn, 1.0f);
    }
  }
}

// ------------------------------------------------------------------------
extern "C" void kernel_launch(void* const* d_in, const int* in_sizes, int n_in,
                              void* d_out, int out_size, void* d_ws,
                              size_t ws_size, hipStream_t stream) {
  const float* x   = (const float*)d_in[0];
  const int*   e0  = (const int*)d_in[1];
  const int*   e1  = (const int*)d_in[2];
  const int*   e2  = (const int*)d_in[3];
  const float* wq0 = (const float*)d_in[5];  const float* bq0 = (const float*)d_in[6];
  const float* wk0 = (const float*)d_in[7];  const float* bk0 = (const float*)d_in[8];
  const float* wv0 = (const float*)d_in[9];  const float* bv0 = (const float*)d_in[10];
  const float* wq1 = (const float*)d_in[11]; const float* bq1 = (const float*)d_in[12];
  const float* wk1 = (const float*)d_in[13]; const float* bk1 = (const float*)d_in[14];
  const float* wv1 = (const float*)d_in[15]; const float* bv1 = (const float*)d_in[16];
  const float* wq2 = (const float*)d_in[17]; const float* bq2 = (const float*)d_in[18];
  const float* wk2 = (const float*)d_in[19]; const float* bk2 = (const float*)d_in[20];
  const float* wv2 = (const float*)d_in[21]; const float* bv2 = (const float*)d_in[22];

  const int N  = in_sizes[0] / 128;   // 50000 nodes
  const int E0 = in_sizes[1] / 2;     // 500000 edges per layer
  const int E1 = in_sizes[2] / 2;
  const int E2 = in_sizes[3] / 2;

  // workspace layout (floats / uints)
  float* x0 = (float*)d_ws;                         // N*128
  float* h1 = x0 + (size_t)N * 128;                 // N*64
  float* sb = h1 + (size_t)N * 64;                  // N*128 (reused per layer)
  float* cb = sb + (size_t)N * 128;                 // N
  unsigned int* wf = (unsigned int*)(cb + N);
  unsigned int* f_q0 = wf;             // 256*128/2 = 16384 uints each
  unsigned int* f_k0 = f_q0 + 16384;
  unsigned int* f_v0 = f_k0 + 16384;
  unsigned int* f_q1 = f_v0 + 16384;   // 256*64/2 = 8192
  unsigned int* f_k1 = f_q1 + 8192;
  unsigned int* f_v1 = f_k1 + 8192;
  unsigned int* f_q2 = f_v1 + 8192;    // 128*128/2 = 8192
  unsigned int* f_k2 = f_q2 + 8192;
  unsigned int* f_v2 = f_k2 + 8192;

  // pack weights into bf16 WMMA B-fragments
  prep_wfrag<<<64, 256, 0, stream>>>(wq0, f_q0, 256, 128);
  prep_wfrag<<<64, 256, 0, stream>>>(wk0, f_k0, 256, 128);
  prep_wfrag<<<64, 256, 0, stream>>>(wv0, f_v0, 256, 128);
  prep_wfrag<<<32, 256, 0, stream>>>(wq1, f_q1, 256, 64);
  prep_wfrag<<<32, 256, 0, stream>>>(wk1, f_k1, 256, 64);
  prep_wfrag<<<32, 256, 0, stream>>>(wv1, f_v1, 256, 64);
  prep_wfrag<<<32, 256, 0, stream>>>(wq2, f_q2, 128, 128);
  prep_wfrag<<<32, 256, 0, stream>>>(wk2, f_k2, 128, 128);
  prep_wfrag<<<32, 256, 0, stream>>>(wv2, f_v2, 128, 128);

  const long tot128 = (long)N * 128;
  const long tot64  = (long)N * 64;
  const int  gz128  = (int)((tot128 + 255) / 256);
  const int  gz64   = (int)((tot64 + 255) / 256);
  const int  gzN    = (N + 255) / 256;

  // ---- layer 0: K=256, O=128 ----
  zero_kernel<<<gz128, 256, 0, stream>>>(sb, tot128);
  zero_kernel<<<gzN, 256, 0, stream>>>(cb, N);
  edgeconv_kernel<256, 128><<<(E0 + 15) / 16, 128, 0, stream>>>(
      x, e0, e0 + E0, f_q0, f_k0, f_v0, bq0, bk0, bv0, sb, cb, E0);
  mean_relu_kernel<<<gz128, 256, 0, stream>>>(sb, cb, x0, 128, tot128);

  // ---- layer 1: K=256, O=64 ----
  zero_kernel<<<gz64, 256, 0, stream>>>(sb, tot64);
  zero_kernel<<<gzN, 256, 0, stream>>>(cb, N);
  edgeconv_kernel<256, 64><<<(E1 + 15) / 16, 128, 0, stream>>>(
      x0, e1, e1 + E1, f_q1, f_k1, f_v1, bq1, bk1, bv1, sb, cb, E1);
  mean_relu_kernel<<<gz64, 256, 0, stream>>>(sb, cb, h1, 64, tot64);

  // ---- layer 2: K=128, O=128, residual + relu into d_out ----
  zero_kernel<<<gz128, 256, 0, stream>>>(sb, tot128);
  zero_kernel<<<gzN, 256, 0, stream>>>(cb, N);
  edgeconv_kernel<128, 128><<<(E2 + 15) / 16, 128, 0, stream>>>(
      h1, e2, e2 + E2, f_q2, f_k2, f_v2, bq2, bk2, bv2, sb, cb, E2);
  mean_add_relu_kernel<<<gz128, 256, 0, stream>>>(sb, cb, x0, (float*)d_out,
                                                  128, tot128);
}
